// Agent_50302656970799
// MI455X (gfx1250) — matrix operand order repaired and checked by reading, not compiled, not run on hardware
//
#include <hip/hip_runtime.h>
#include <hip/hip_bf16.h>
#include <cstddef>
#include <cstdint>

// ---------------------------------------------------------------------------
// Problem constants (match reference)
// ---------------------------------------------------------------------------
#define F_DIM 128
#define H_DIM 64
#define A_DIM 15
#define T_DIM 128
#define B_DIM 2048

typedef __attribute__((ext_vector_type(16))) _Float16 v16h;
typedef __attribute__((ext_vector_type(8)))  float    v8f;

// ---------------------------------------------------------------------------
// WMMA helper (CDNA5 v_wmma_f32_16x16x32_f16)
// ---------------------------------------------------------------------------
__device__ __forceinline__ v8f wmma_f16(v16h a, v16h b, v8f c) {
    return __builtin_amdgcn_wmma_f32_16x16x32_f16(
        false, a, false, b, (short)0, c, false, false);
}

// Branch-free activations: v_exp_f32 + v_rcp_f32, no EXEC divergence.
// sigmoid(+inf)=1, sigmoid(-inf)=0; tanh(+-inf)=+-1 (exp saturates to inf/0).
__device__ __forceinline__ float sigmoid_fast(float x) {
    return __builtin_amdgcn_rcpf(1.0f + __expf(-x));
}
__device__ __forceinline__ float tanh_fast(float x) {
    return 2.0f * __builtin_amdgcn_rcpf(1.0f + __expf(-2.0f * x)) - 1.0f;
}

// ---------------------------------------------------------------------------
// gfx1250 async global->LDS copy (ASYNCcnt-tracked), inline asm per
// CDNA5 ISA (portable across both toolchains).
// Per-lane: 8 bytes global -> 8 bytes LDS.
// ---------------------------------------------------------------------------
__device__ __forceinline__ void async_load_b64_to_lds(void* lds_dst,
                                                      const void* gsrc) {
    // Low 32 bits of the generic pointer are the LDS byte offset (aperture mapping).
    uint32_t ldsoff = (uint32_t)(uintptr_t)lds_dst;
    asm volatile("global_load_async_to_lds_b64 %0, %1, off"
                 :: "v"(ldsoff), "v"(gsrc)
                 : "memory");
}
__device__ __forceinline__ void wait_asynccnt0() {
    asm volatile("s_wait_asynccnt 0" ::: "memory");
}
// Compiler-only scheduling fence for wave-internal cross-lane LDS deps
// (hardware keeps same-wave LDS ops in order).
__device__ __forceinline__ void sched_fence() {
    asm volatile("" ::: "memory");
}

// ---------------------------------------------------------------------------
// Fragment loaders
// ---------------------------------------------------------------------------
// A-matrix 16x32 f16 fragment from row-major f16 (LDS).
// Lane l: M=l&15.  Halves j: K = j + (l>=16?16:0).
// Compiles to 2x ds_load_b128 per fragment.
__device__ __forceinline__ v16h load_A16x32(const _Float16* __restrict__ src,
                                            int ld, int koff) {
    const int lane = threadIdx.x & 31;
    const int m = lane & 15, hi = lane >> 4;
    const _Float16* row = src + (size_t)m * ld + koff + hi * 16;
    v16h a;
#pragma unroll
    for (int v = 0; v < 8; ++v) {
        const int kb = 2 * v;
        a[2 * v]     = row[kb];
        a[2 * v + 1] = row[kb + 1];
    }
    return a;
}

// B-matrix 32x16 f16 fragment from N-major f16 storage Wt[n][k].
__device__ __forceinline__ v16h load_B32x16_lds(const _Float16* __restrict__ Wt,
                                                int ld, int ntile, int kstep) {
    const int lane = threadIdx.x & 31;
    const int n = lane & 15, hi = lane >> 4;
    const _Float16* p = Wt + (size_t)(ntile * 16 + n) * ld + kstep * 32 + hi * 16;
    v16h b;
#pragma unroll
    for (int j = 0; j < 16; ++j) b[j] = p[j];
    return b;
}

// Same fragment, converting from global f32 N-major weights (Wih/Whh are
// [4H, H] row-major and g = y @ W^T, so row n is K-contiguous).
__device__ __forceinline__ v16h load_B32x16_gf32(const float* __restrict__ W,
                                                 int ld, int ntile, int kstep) {
    const int lane = threadIdx.x & 31;
    const int n = lane & 15, hi = lane >> 4;
    const float* p = W + (size_t)(ntile * 16 + n) * ld + kstep * 32 + hi * 16;
    v16h b;
#pragma unroll
    for (int j = 0; j < 16; ++j) b[j] = (_Float16)p[j];
    return b;
}

// ---------------------------------------------------------------------------
// Kernel 1: MLP encoder  y = tanh(tanh(x@W1+b1)@W2+b2)  -> f16 workspace
// 256 threads (8 waves), 128 rows per block.  24 WMMA / wave.
// ---------------------------------------------------------------------------
__global__ __launch_bounds__(256) void encoder_kernel(
    const float* __restrict__ x,
    const float* __restrict__ W1, const float* __restrict__ b1,
    const float* __restrict__ W2, const float* __restrict__ b2,
    _Float16* __restrict__ yenc)
{
    __shared__ __align__(16) _Float16 sX[128][136];   // x tile (reused for y1/y2)
    __shared__ __align__(16) _Float16 sW1t[64][136];  // W1^T: [n][k], K contiguous
    __shared__ __align__(16) _Float16 sW2t[64][72];   // W2^T
    __shared__ float sB1[64], sB2[64];

    const int tid = threadIdx.x;
    const int rowBase = blockIdx.x * 128;

    // Stage W1/W2 transposed to N-major f16 (coalesced global reads).
    for (int i = tid; i < F_DIM * H_DIM; i += 256) {
        const int k = i >> 6, n = i & 63;
        sW1t[n][k] = (_Float16)W1[i];
    }
    for (int i = tid; i < H_DIM * H_DIM; i += 256) {
        const int k = i >> 6, n = i & 63;
        sW2t[n][k] = (_Float16)W2[i];
    }
    if (tid < 64) { sB1[tid] = b1[tid]; sB2[tid] = b2[tid]; }

    // Stage x tile f32 -> f16 with b128 global loads (32 float4 per row).
    {
        const float4* xv = (const float4*)(x + (size_t)rowBase * F_DIM);
        for (int i = tid; i < 128 * F_DIM / 4; i += 256) {
            const float4 v = xv[i];
            const int r = i >> 5, c = (i & 31) * 4;
            sX[r][c + 0] = (_Float16)v.x;
            sX[r][c + 1] = (_Float16)v.y;
            sX[r][c + 2] = (_Float16)v.z;
            sX[r][c + 3] = (_Float16)v.w;
        }
    }
    __syncthreads();

    const int wave = tid >> 5;
    const int lane = tid & 31;
    const int m0 = wave * 16;             // wave's 16-row strip
    const int n = lane & 15, hi = lane >> 4;

    // Preload all 4 K-step A fragments for this wave's rows.
    v16h a1[4];
#pragma unroll
    for (int s = 0; s < 4; ++s) a1[s] = load_A16x32(&sX[m0][0], 136, s * 32);
    __syncthreads();   // everyone done reading sX before y1 overlays it

    _Float16 (*sY)[72] = reinterpret_cast<_Float16(*)[72]>(&sX[0][0]);

    // GEMM1: 16x64 += 16x128 @ 128x64, tanh -> sY (wave writes only own rows)
#pragma unroll
    for (int nt = 0; nt < 4; ++nt) {
        v8f acc = {};
#pragma unroll
        for (int s = 0; s < 4; ++s)
            acc = wmma_f16(a1[s], load_B32x16_lds(&sW1t[0][0], 136, nt, s), acc);
        const int col = nt * 16 + n;
#pragma unroll
        for (int v = 0; v < 8; ++v)
            sY[m0 + v + hi * 8][col] = (_Float16)tanh_fast(acc[v] + sB1[col]);
    }
    __syncthreads();

    // GEMM2: 16x64 += 16x64 @ 64x64, tanh -> sY in place (same row ownership)
    v16h a2[2];
#pragma unroll
    for (int s = 0; s < 2; ++s) a2[s] = load_A16x32(&sY[m0][0], 72, s * 32);

#pragma unroll
    for (int nt = 0; nt < 4; ++nt) {
        v8f acc = {};
#pragma unroll
        for (int s = 0; s < 2; ++s)
            acc = wmma_f16(a2[s], load_B32x16_lds(&sW2t[0][0], 72, nt, s), acc);
        const int col = nt * 16 + n;
#pragma unroll
        for (int v = 0; v < 8; ++v)
            sY[m0 + v + hi * 8][col] = (_Float16)tanh_fast(acc[v] + sB2[col]);
    }
    __syncthreads();

    // Coalesced copy-out: thread t moves half a row (32 halves = 4x b128).
    {
        const int r = tid >> 1, c = (tid & 1) * 32;
        const uint4* src = (const uint4*)&sY[r][c];
        uint4* dst = (uint4*)(yenc + (size_t)(rowBase + r) * H_DIM + c);
        dst[0] = src[0];
        dst[1] = src[1];
        dst[2] = src[2];
        dst[3] = src[3];
    }
}

// ---------------------------------------------------------------------------
// Kernel 2: LSTM scan (T=128) + fused actor/critic heads.
// Recurrence mixes only H -> partition B across 128 independent blocks of
// 16 rows.  h/c resident in LDS, Wih/Whh resident in registers, per-step
// yt tile (2 KB, contiguous) async-prefetched into a double buffer.
// Per step per wave: 8 v_wmma.  2 barriers per step.
// ---------------------------------------------------------------------------
__global__ __launch_bounds__(256) void lstm_head_kernel(
    const _Float16* __restrict__ yenc, const float* __restrict__ done,
    const float* __restrict__ h0, const float* __restrict__ c0,
    const float* __restrict__ Wih, const float* __restrict__ bih,
    const float* __restrict__ Whh, const float* __restrict__ bhh,
    const float* __restrict__ Wa, const float* __restrict__ ba,
    const float* __restrict__ Wc, const float* __restrict__ bc,
    float* __restrict__ out)
{
    __shared__ __align__(16) _Float16 sY[2][16][64];  // async yt double buffer
    __shared__ __align__(16) _Float16 sH[16][72];     // hidden state (f16)
    __shared__ float sC[16][64];                      // cell state (f32)
    __shared__ float sG[16][264];                     // gate pre-activations
    __shared__ float sBias[256];                      // bih + bhh
    __shared__ float sWheadT[16][68];                 // rows 0..14 Wa^T, 15 Wc^T
    __shared__ float sBhead[16];

    const int tid = threadIdx.x;
    const int wave = tid >> 5;
    const int lane = tid & 31;
    const int n = lane & 15, hi = lane >> 4;
    const int rowBase = blockIdx.x * 16;              // batch rows owned

    sBias[tid] = bih[tid] + bhh[tid];
    for (int i = tid; i < 64 * 16; i += 256) {
        const int k = i >> 4, cidx = i & 15;
        sWheadT[cidx][k] = (cidx < A_DIM) ? Wa[(size_t)k * A_DIM + cidx] : Wc[k];
    }
    if (tid < 16) sBhead[tid] = (tid < A_DIM) ? ba[tid] : bc[0];
    for (int i = tid; i < 16 * 64; i += 256) {
        const int r = i >> 6, k = i & 63;
        sH[r][k] = (_Float16)h0[(size_t)(rowBase + r) * H_DIM + k];
        sC[r][k] = c0[(size_t)(rowBase + r) * H_DIM + k];
    }

    // Resident recurrent weight fragments: wave owns N-tiles {2w, 2w+1}.
    v16h Bih[2][2], Bhh[2][2];
#pragma unroll
    for (int j = 0; j < 2; ++j) {
        const int ntg = wave * 2 + j;
#pragma unroll
        for (int s = 0; s < 2; ++s) {
            Bih[j][s] = load_B32x16_gf32(Wih, H_DIM, ntg, s);
            Bhh[j][s] = load_B32x16_gf32(Whh, H_DIM, ntg, s);
        }
    }

    // Kick off async prefetch of the t=0 tile (256 threads x 8B = 2 KB).
    async_load_b64_to_lds(&sY[0][0][0] + tid * 4,
                          yenc + (size_t)rowBase * H_DIM + tid * 4);
    __syncthreads();

    const size_t vf_off = (size_t)T_DIM * B_DIM * A_DIM;
    const size_t hT_off = vf_off + (size_t)T_DIM * B_DIM;
    const size_t cT_off = hT_off + (size_t)B_DIM * H_DIM;

    // Elementwise phase mapping: thread owns row r = tid/16 (wave-local rows
    // 2w, 2w+1), k0 = (tid%16)*4 .. +3.  Head phase uses the same rows.
    const int er = tid >> 4;
    const int ek = (tid & 15) * 4;

    for (int t = 0; t < T_DIM; ++t) {
        wait_asynccnt0();                 // this wave's yt chunk landed

        // ---- done-mask reset of h and c (own-wave rows only) ----
        {
            const float m = 1.0f - done[(size_t)t * B_DIM + rowBase + er];
#pragma unroll
            for (int e = 0; e < 4; ++e) {
                sH[er][ek + e] = (_Float16)((float)sH[er][ek + e] * m);
                sC[er][ek + e] *= m;
            }
        }
        __syncthreads();                  // all yt chunks + all masks visible

        // ---- issue async prefetch of next yt tile into the other buffer ----
        if (t + 1 < T_DIM) {
            async_load_b64_to_lds(
                &sY[(t + 1) & 1][0][0] + tid * 4,
                yenc + ((size_t)(t + 1) * B_DIM + rowBase) * H_DIM + tid * 4);
        }

        // ---- gate GEMM: g = yt@Wih^T + h@Whh^T + (bih+bhh) ----
        const _Float16* yt = &sY[t & 1][0][0];
        const v16h Ay0 = load_A16x32(yt, 64, 0);
        const v16h Ay1 = load_A16x32(yt, 64, 32);
        const v16h Ah0 = load_A16x32(&sH[0][0], 72, 0);
        const v16h Ah1 = load_A16x32(&sH[0][0], 72, 32);
#pragma unroll
        for (int j = 0; j < 2; ++j) {
            const int ntg = wave * 2 + j;
            v8f acc = {};
            acc = wmma_f16(Ay0, Bih[j][0], acc);
            acc = wmma_f16(Ay1, Bih[j][1], acc);
            acc = wmma_f16(Ah0, Bhh[j][0], acc);
            acc = wmma_f16(Ah1, Bhh[j][1], acc);
            const int col = ntg * 16 + n;
#pragma unroll
            for (int v = 0; v < 8; ++v)
                sG[v + hi * 8][col] = acc[v] + sBias[col];
        }
        __syncthreads();                  // sG complete

        // ---- LSTM cell update (gate order i,f,g,o), branch-free ----
#pragma unroll
        for (int e = 0; e < 4; ++e) {
            const int k = ek + e;
            const float gi = sigmoid_fast(sG[er][k]);
            const float gf = sigmoid_fast(sG[er][64 + k]);
            const float gg = tanh_fast(sG[er][128 + k]);
            const float go = sigmoid_fast(sG[er][192 + k]);
            const float c  = gf * sC[er][k] + gi * gg;
            sC[er][k] = c;
            sH[er][k] = (_Float16)(go * tanh_fast(c));
        }
        sched_fence();   // wave-internal: heads read sH written by sibling lanes
                         // (same-wave LDS ops are processed in order)

        // ---- fused heads: one output element per thread (15 logits + vf),
        //      reads only this wave's rows of sH -> no barrier needed ----
        {
            const int cidx = tid & 15;
            float dot = sBhead[cidx];
#pragma unroll
            for (int k = 0; k < 64; ++k)
                dot += (float)sH[er][k] * sWheadT[cidx][k];
            const size_t grow = (size_t)t * B_DIM + rowBase + er;
            if (cidx < A_DIM) out[grow * A_DIM + cidx] = dot;
            else              out[vf_off + grow] = dot;
        }
        // No end-of-step barrier: next mask touches only own-wave rows; the
        // barrier after the mask orders cross-wave sH/sG reuse.
    }

    // ---- final hT / cT ----
#pragma unroll
    for (int e = 0; e < 4; ++e) {
        const int k = ek + e;
        out[hT_off + (size_t)(rowBase + er) * H_DIM + k] = (float)sH[er][k];
        out[cT_off + (size_t)(rowBase + er) * H_DIM + k] = sC[er][k];
    }
}

// ---------------------------------------------------------------------------
// Launch: encoder (2048 blocks) then LSTM+heads (128 independent B-slices).
// Workspace: y_enc = [T*B, H] f16 = 32 MiB.
// ---------------------------------------------------------------------------
extern "C" void kernel_launch(void* const* d_in, const int* in_sizes, int n_in,
                              void* d_out, int out_size, void* d_ws, size_t ws_size,
                              hipStream_t stream) {
    const float* x    = (const float*)d_in[0];
    const float* done = (const float*)d_in[1];
    const float* h0   = (const float*)d_in[2];
    const float* c0   = (const float*)d_in[3];
    const float* W1   = (const float*)d_in[4];
    const float* b1   = (const float*)d_in[5];
    const float* W2   = (const float*)d_in[6];
    const float* b2   = (const float*)d_in[7];
    const float* Wih  = (const float*)d_in[8];
    const float* bih  = (const float*)d_in[9];
    const float* Whh  = (const float*)d_in[10];
    const float* bhh  = (const float*)d_in[11];
    const float* Wa   = (const float*)d_in[12];
    const float* ba   = (const float*)d_in[13];
    const float* Wc   = (const float*)d_in[14];
    const float* bc   = (const float*)d_in[15];

    _Float16* yenc = (_Float16*)d_ws;       // [T*B, H] f16
    float* out = (float*)d_out;

    encoder_kernel<<<(T_DIM * B_DIM) / 128, 256, 0, stream>>>(x, W1, b1, W2, b2, yenc);
    lstm_head_kernel<<<B_DIM / 16, 256, 0, stream>>>(
        yenc, done, h0, c0, Wih, bih, Whh, bhh, Wa, ba, Wc, bc, out);
}
